// FBSparseVGGTest_33809982554780
// MI455X (gfx1250) — matrix operand order repaired and checked by reading, not compiled, not run on hardware
//
#include <hip/hip_runtime.h>
#include <hip/hip_bf16.h>
#include <stdint.h>

typedef __attribute__((ext_vector_type(16))) _Float16 v16h;
typedef __attribute__((ext_vector_type(8)))  _Float16 v8h;
typedef __attribute__((ext_vector_type(8)))  float    v8f;
typedef __attribute__((ext_vector_type(4)))  int      v4i;

#define BSZ 32
#define H0 199
#define W0 317
#define NEVT 262144
#define NCLS 101
#define BNEPS 1e-4f

#define AS1 __attribute__((address_space(1)))
#define AS3 __attribute__((address_space(3)))

#if defined(__HIP_DEVICE_COMPILE__)
#if __has_builtin(__builtin_amdgcn_global_load_async_to_lds_b128)
#define USE_ASYNC_LDS 1
#endif
#endif

__device__ __forceinline__ AS1 v4i* to_global_as(const void* p) {
  return (AS1 v4i*)(uintptr_t)p;
}
__device__ __forceinline__ AS3 v4i* to_shared_as(const void* p) {
  // flat LDS address: low 32 bits == LDS offset (ISA aperture rule)
  return (AS3 v4i*)(unsigned)(uintptr_t)p;
}

template <int N>
__device__ __forceinline__ void async_wait_le() {
#ifdef USE_ASYNC_LDS
#if __has_builtin(__builtin_amdgcn_s_wait_asynccnt)
  __builtin_amdgcn_s_wait_asynccnt((unsigned short)N);
#else
  asm volatile("s_wait_asynccnt %0" ::"n"(N) : "memory");
#endif
#endif
}

// ---------------- utility kernels ----------------

__global__ void k_zero_f32(float* __restrict__ p, long n) {
  long i = (long)blockIdx.x * blockDim.x + threadIdx.x;
  if (i < n) p[i] = 0.f;
}

__global__ void k_scatter(const int* __restrict__ coords, const float* __restrict__ feats,
                          float* __restrict__ x0, float* __restrict__ m0, int nev) {
  int i = blockIdx.x * blockDim.x + threadIdx.x;
  if (i >= nev) return;
  int y = coords[3 * i + 0];
  int x = coords[3 * i + 1];
  int b = coords[3 * i + 2];
  long pix = ((long)b * H0 + y) * W0 + x;
  atomicAdd(&x0[pix * 2 + 0], feats[2 * i + 0]);
  atomicAdd(&x0[pix * 2 + 1], feats[2 * i + 1]);
  m0[pix] = 1.0f;
}

__global__ void k_f32_to_f16(const float* __restrict__ in, _Float16* __restrict__ out, long n) {
  long i = (long)blockIdx.x * blockDim.x + threadIdx.x;
  if (i < n) out[i] = (_Float16)in[i];
}

__global__ void k_bn_prep(const float* __restrict__ sc, const float* __restrict__ bi,
                          const float* __restrict__ me, const float* __restrict__ va,
                          float* __restrict__ inv, float* __restrict__ beta, int C) {
  int i = blockIdx.x * blockDim.x + threadIdx.x;
  if (i >= C) return;
  float iv = sc[i] * rsqrtf(va[i] + BNEPS);
  inv[i] = iv;
  beta[i] = bi[i] - me[i] * iv;
}

// Repack HWIO f32 weights into the WMMA B-matrix (32x16 f16 per tile) lane layout:
// tile (kt, nt): lane l (0..31), element e (0..15):
//   k = kt*32 + (l>=16 ? 16 : 0) + e ; n = nt*16 + (l&15)
__global__ void k_pack_w(const float* __restrict__ w, _Float16* __restrict__ wp,
                         int K, int Cout, int Ktiles) {
  long total = (long)Ktiles * Cout * 32;
  long idx = (long)blockIdx.x * blockDim.x + threadIdx.x;
  if (idx >= total) return;
  int e = (int)(idx & 15);
  int l = (int)((idx >> 4) & 31);
  long t = idx >> 9;
  int nt16 = Cout >> 4;
  int nt = (int)(t % nt16);
  int kt = (int)(t / nt16);
  int k = kt * 32 + ((l >= 16) ? 16 : 0) + e;
  int n = nt * 16 + (l & 15);
  wp[idx] = (k < K) ? (_Float16)w[(long)k * Cout + n] : (_Float16)0.f;
}

// ---------------- WMMA conv kernel ----------------
// 128-thread blocks = 4 waves; each wave computes a 16(pixels) x NB*16(out-channels)
// tile, reusing one A tile across NB WMMAs per K-chunk of 32.
// B (weight) tiles are shared by all 4 waves -> staged once per block into LDS,
// double-buffered with async global->LDS copies overlapped with WMMA compute.
// FAST requires Cin % 8 == 0: every aligned 8-wide K-run stays inside one 3x3 tap,
// so A is built from two 128-bit loads per lane.
template <int NB, bool FAST>
__global__ __launch_bounds__(128) void k_conv_wmma(
    const _Float16* __restrict__ in, _Float16* __restrict__ out,
    const _Float16* __restrict__ wpack,
    const float* __restrict__ bnInv, const float* __restrict__ bnBeta,
    const float* __restrict__ mask,
    int Hin, int Win, int Cin, int Hout, int Wout, int Cout,
    int stride, int pad, int cinShift, int K, int Ktiles)
{
  const int lane = threadIdx.x & 31;
  const int wid  = threadIdx.x >> 5;
  const int hi   = lane >> 4;      // K-half select per A/B lane layout
  const int mrow = lane & 15;
  const long p0 = ((long)blockIdx.x * 4 + wid) * 16;
  const int ntB = blockIdx.y * NB;
  const int NT = Cout >> 4;
  const int totalP = BSZ * Hout * Wout;

  long p = p0 + mrow;
  bool pvalid = p < totalP;
  int tp = pvalid ? (int)p : 0;
  int ox = tp % Wout; tp /= Wout;
  int oy = tp % Hout;
  int b  = tp / Hout;

  __shared__ _Float16 ldsB[2][NB * 512];
  constexpr int CHUNKS = NB * 64;                  // 16-byte chunks per stage
  constexpr int STAGE_OPS = (CHUNKS + 127) / 128;  // async ops per wave per stage

  auto issue_stage = [&](int kt, int buf) {
    const _Float16* g0 = wpack + ((long)kt * NT + ntB) * 512;
#ifdef USE_ASYNC_LDS
    for (int c = threadIdx.x; c < CHUNKS; c += 128) {
      const _Float16* g = g0 + c * 8;
      _Float16* l = &ldsB[buf][c * 8];
      __builtin_amdgcn_global_load_async_to_lds_b128(
          to_global_as(g), to_shared_as(l), 0, 0);
    }
#else
    const unsigned long long* gq = (const unsigned long long*)g0;
    unsigned long long* lq = (unsigned long long*)&ldsB[buf][0];
#pragma unroll
    for (int c = 0; c < NB; ++c)
      lq[threadIdx.x + c * 128] = gq[threadIdx.x + c * 128];
#endif
  };

  v8f acc[NB];
#pragma unroll
  for (int i = 0; i < NB; ++i) acc[i] = (v8f){};

  issue_stage(0, 0);

  for (int kt = 0; kt < Ktiles; ++kt) {
    const int buf = kt & 1;
    const bool more = (kt + 1) < Ktiles;
    if (more) issue_stage(kt + 1, buf ^ 1);
    if (more) async_wait_le<STAGE_OPS>();
    else      async_wait_le<0>();
    __syncthreads();   // stage buf visible to all waves

    const int kk = kt * 32;
    v16h A;
    if (FAST) {
      v8h lo = {}, hp = {};
      {
        int k0 = kk + (hi << 4);                       // A elements 0..7
        if (k0 < K) {
          int tap = k0 >> cinShift, ci = k0 & (Cin - 1);
          int ky = tap / 3, kx = tap - ky * 3;
          int iy = oy * stride + ky - pad, ix = ox * stride + kx - pad;
          if (pvalid && (unsigned)iy < (unsigned)Hin && (unsigned)ix < (unsigned)Win) {
            const _Float16* src = in + (((long)(b * Hin + iy) * Win + ix) << cinShift) + ci;
            lo = *(const v8h*)src;
            __builtin_prefetch((const void*)(src + 32), 0, 0);  // next K-chunk stream
          }
        }
      }
      {
        int k0 = kk + 8 + (hi << 4);                   // A elements 8..15
        if (k0 < K) {
          int tap = k0 >> cinShift, ci = k0 & (Cin - 1);
          int ky = tap / 3, kx = tap - ky * 3;
          int iy = oy * stride + ky - pad, ix = ox * stride + kx - pad;
          if (pvalid && (unsigned)iy < (unsigned)Hin && (unsigned)ix < (unsigned)Win) {
            const _Float16* src = in + (((long)(b * Hin + iy) * Win + ix) << cinShift) + ci;
            hp = *(const v8h*)src;
          }
        }
      }
#pragma unroll
      for (int e = 0; e < 8; ++e) { A[e] = lo[e]; A[e + 8] = hp[e]; }
    } else {
      // generic gather path (Cin = 2)
#pragma unroll
      for (int e = 0; e < 16; ++e) {
        int k = kk + e + (hi << 4);
        _Float16 v = (_Float16)0.f;
        if (pvalid && k < K) {
          int tap = k >> cinShift;
          int ci = k & (Cin - 1);
          int ky = tap / 3, kx = tap - ky * 3;
          int iy = oy * stride + ky - pad;
          int ix = ox * stride + kx - pad;
          if ((unsigned)iy < (unsigned)Hin && (unsigned)ix < (unsigned)Win)
            v = in[(((long)(b * Hin + iy) * Win + ix) << cinShift) + ci];
        }
        A[e] = v;
      }
    }

#pragma unroll
    for (int nb = 0; nb < NB; ++nb) {
      v16h Bm = *(const v16h*)&ldsB[buf][nb * 512 + lane * 16];  // ds_load, 32B
      acc[nb] = __builtin_amdgcn_wmma_f32_16x16x32_f16(
          /*neg_a=*/false, A, /*neg_b=*/false, Bm,
          /*c_mod=*/(short)0, acc[nb], /*reuse_a=*/false, /*reuse_b=*/false);
    }
    __syncthreads();   // all waves done reading buf before it is overwritten
  }

  // Epilogue: fused BN + ReLU + mask, store f16 NHWC.
  // C/D layout: VGPR r -> M = r + hi*8 ; N = lane&15.
  long pixArr[8];
  float mv[8];
  bool okArr[8];
#pragma unroll
  for (int r = 0; r < 8; ++r) {
    long pe = p0 + r + (hi << 3);
    bool ok = pe < totalP;
    okArr[r] = ok;
    int te = ok ? (int)pe : 0;
    int ex = te % Wout; te /= Wout;
    int ey = te % Hout;
    int eb = te / Hout;
    long pix = ((long)eb * Hout + ey) * Wout + ex;
    pixArr[r] = pix;
    mv[r] = ok ? mask[pix] : 0.f;
  }
#pragma unroll
  for (int nb = 0; nb < NB; ++nb) {
    const int n = (ntB + nb) * 16 + mrow;
    const float inv  = bnInv[n];
    const float beta = bnBeta[n];
#pragma unroll
    for (int r = 0; r < 8; ++r) {
      if (okArr[r]) {
        float v = acc[nb][r] * inv + beta;
        v = fmaxf(v, 0.f) * mv[r];
        out[pixArr[r] * (long)Cout + n] = (_Float16)v;
      }
    }
  }
}

// ---------------- pooling ----------------

__global__ void k_maxpool_f16(const _Float16* __restrict__ in, _Float16* __restrict__ out,
                              int Hin, int Win, int Ho, int Wo, int C) {
  long total = (long)BSZ * Ho * Wo * C;
  long idx = (long)blockIdx.x * blockDim.x + threadIdx.x;
  if (idx >= total) return;
  int c = (int)(idx % C); long t = idx / C;
  int ox = (int)(t % Wo); t /= Wo;
  int oy = (int)(t % Ho);
  int b  = (int)(t / Ho);
  float m = -1e30f;
#pragma unroll
  for (int ky = 0; ky < 3; ++ky)
#pragma unroll
    for (int kx = 0; kx < 3; ++kx) {
      int iy = oy * 2 + ky, ix = ox * 2 + kx;
      if (iy < Hin && ix < Win)
        m = fmaxf(m, (float)in[(((long)b * Hin + iy) * Win + ix) * C + c]);
    }
  out[idx] = (_Float16)m;
}

__global__ void k_maxpool_f32(const float* __restrict__ in, float* __restrict__ out,
                              int Hin, int Win, int Ho, int Wo) {
  long total = (long)BSZ * Ho * Wo;
  long idx = (long)blockIdx.x * blockDim.x + threadIdx.x;
  if (idx >= total) return;
  long t = idx;
  int ox = (int)(t % Wo); t /= Wo;
  int oy = (int)(t % Ho);
  int b  = (int)(t / Ho);
  float m = -1e30f;
#pragma unroll
  for (int ky = 0; ky < 3; ++ky)
#pragma unroll
    for (int kx = 0; kx < 3; ++kx) {
      int iy = oy * 2 + ky, ix = ox * 2 + kx;
      if (iy < Hin && ix < Win)
        m = fmaxf(m, in[((long)b * Hin + iy) * Win + ix]);
    }
  out[idx] = m;
}

// ---------------- final linear ----------------
// act: (B,2,3,256) f16 NHWC; feature f = c*6 + y*3 + x (matches transpose+reshape)
__global__ void k_linear(const _Float16* __restrict__ act, const float* __restrict__ lw,
                         const float* __restrict__ lb, float* __restrict__ out) {
  int b = blockIdx.x;
  int j = threadIdx.x;
  if (j >= NCLS) return;
  float s = lb[j];
  for (int f = 0; f < 1536; ++f) {
    int c = f / 6;
    int yx = f - c * 6;
    int y = yx / 3, x = yx - (yx / 3) * 3;
    s += (float)act[(((long)b * 2 + y) * 3 + x) * 256 + c] * lw[(long)j * 1536 + f];
  }
  out[b * NCLS + j] = s;
}

// ---------------- host launcher ----------------

extern "C" void kernel_launch(void* const* d_in, const int* in_sizes, int n_in,
                              void* d_out, int out_size, void* d_ws, size_t ws_size,
                              hipStream_t stream) {
  (void)in_sizes; (void)n_in; (void)out_size; (void)ws_size;

  const int*   coords = (const int*)d_in[0];
  const float* feats  = (const float*)d_in[1];
  const float* convw[12];
  const float* bnS[12]; const float* bnB[12]; const float* bnM[12]; const float* bnV[12];
  for (int i = 0; i < 12; ++i) {
    convw[i] = (const float*)d_in[2 + i];
    bnS[i]   = (const float*)d_in[14 + i];
    bnB[i]   = (const float*)d_in[26 + i];
    bnM[i]   = (const float*)d_in[38 + i];
    bnV[i]   = (const float*)d_in[50 + i];
  }
  const float* linw = (const float*)d_in[62];
  const float* linb = (const float*)d_in[63];
  float* outp = (float*)d_out;

  const int HH[7] = {199, 99, 49, 24, 11, 5, 2};
  const int WW[7] = {317, 158, 78, 38, 18, 8, 3};
  struct LayerCfg { int Cin, Cout, lin, lout, stride, pad; };
  const LayerCfg LC[12] = {
    {2, 16, 0, 0, 1, 1},   {16, 16, 0, 0, 1, 1},
    {16, 32, 1, 1, 1, 1},  {32, 32, 1, 1, 1, 1},
    {32, 64, 2, 2, 1, 1},  {64, 64, 2, 2, 1, 1},
    {64, 128, 3, 3, 1, 1}, {128, 128, 3, 3, 1, 1},
    {128, 256, 4, 4, 1, 1},{256, 256, 4, 4, 1, 1},
    {256, 512, 5, 5, 1, 1},
    {512, 256, 5, 6, 2, 0},
  };

  // ---- workspace carve ----
  size_t off = 0;
  auto carve = [&](size_t bytes) -> void* {
    void* p = (char*)d_ws + off;
    off = (off + bytes + 255) & ~(size_t)255;
    return p;
  };
  float* x0 = (float*)carve((size_t)BSZ * H0 * W0 * 2 * sizeof(float));
  float* mptr[7];
  for (int l = 0; l < 7; ++l)
    mptr[l] = (float*)carve((size_t)BSZ * HH[l] * WW[l] * sizeof(float));
  const size_t maxActHalfs = (size_t)BSZ * H0 * W0 * 16;  // largest f16 activation
  _Float16* actA = (_Float16*)carve(maxActHalfs * sizeof(_Float16));
  _Float16* actB = (_Float16*)carve(maxActHalfs * sizeof(_Float16));
  _Float16* wp[12];
  float* bnInvP[12]; float* bnBetaP[12];
  for (int li = 0; li < 12; ++li) {
    int K = 9 * LC[li].Cin;
    int Kt = (K + 31) / 32;
    wp[li] = (_Float16*)carve((size_t)Kt * LC[li].Cout * 32 * sizeof(_Float16));
    bnInvP[li]  = (float*)carve((size_t)LC[li].Cout * sizeof(float));
    bnBetaP[li] = (float*)carve((size_t)LC[li].Cout * sizeof(float));
  }

  // ---- preprocessing ----
  {
    long nx = (long)BSZ * H0 * W0 * 2;
    long nm = (long)BSZ * H0 * W0;
    k_zero_f32<<<(unsigned)((nx + 255) / 256), 256, 0, stream>>>(x0, nx);
    k_zero_f32<<<(unsigned)((nm + 255) / 256), 256, 0, stream>>>(mptr[0], nm);
    k_scatter<<<(NEVT + 255) / 256, 256, 0, stream>>>(coords, feats, x0, mptr[0], NEVT);
    k_f32_to_f16<<<(unsigned)((nx + 255) / 256), 256, 0, stream>>>(x0, actB, nx);
  }
  for (int li = 0; li < 12; ++li) {
    int K = 9 * LC[li].Cin;
    int Kt = (K + 31) / 32;
    long tot = (long)Kt * LC[li].Cout * 32;
    k_pack_w<<<(unsigned)((tot + 255) / 256), 256, 0, stream>>>(convw[li], wp[li], K, LC[li].Cout, Kt);
    int C = LC[li].Cout;
    k_bn_prep<<<(C + 63) / 64, 64, 0, stream>>>(bnS[li], bnB[li], bnM[li], bnV[li],
                                                bnInvP[li], bnBetaP[li], C);
  }
  // mask pyramid
  for (int l = 0; l < 6; ++l) {
    long tot = (long)BSZ * HH[l + 1] * WW[l + 1];
    k_maxpool_f32<<<(unsigned)((tot + 255) / 256), 256, 0, stream>>>(
        mptr[l], mptr[l + 1], HH[l], WW[l], HH[l + 1], WW[l + 1]);
  }

  // ---- conv / pool pipeline (bufB = current input, bufA = next output) ----
  _Float16* bufA = actA;
  _Float16* bufB = actB;
  auto conv = [&](int li) {
    const LayerCfg& c = LC[li];
    int Hin = HH[c.lin], Win = WW[c.lin], Hout = HH[c.lout], Wout = WW[c.lout];
    int K = 9 * c.Cin, Kt = (K + 31) / 32, sh = __builtin_ctz((unsigned)c.Cin);
    int totalP = BSZ * Hout * Wout;
    int Mtiles = (totalP + 15) / 16;
    int NT = c.Cout / 16;
    int NB = (NT >= 4) ? 4 : NT;
    dim3 g((unsigned)((Mtiles + 3) / 4), (unsigned)(NT / NB));
    bool fast = (c.Cin % 8) == 0;
    if (fast) {
      if (NB == 4)
        k_conv_wmma<4, true><<<g, 128, 0, stream>>>(bufB, bufA, wp[li], bnInvP[li], bnBetaP[li],
            mptr[c.lout], Hin, Win, c.Cin, Hout, Wout, c.Cout, c.stride, c.pad, sh, K, Kt);
      else if (NB == 2)
        k_conv_wmma<2, true><<<g, 128, 0, stream>>>(bufB, bufA, wp[li], bnInvP[li], bnBetaP[li],
            mptr[c.lout], Hin, Win, c.Cin, Hout, Wout, c.Cout, c.stride, c.pad, sh, K, Kt);
      else
        k_conv_wmma<1, true><<<g, 128, 0, stream>>>(bufB, bufA, wp[li], bnInvP[li], bnBetaP[li],
            mptr[c.lout], Hin, Win, c.Cin, Hout, Wout, c.Cout, c.stride, c.pad, sh, K, Kt);
    } else {
      k_conv_wmma<1, false><<<g, 128, 0, stream>>>(bufB, bufA, wp[li], bnInvP[li], bnBetaP[li],
          mptr[c.lout], Hin, Win, c.Cin, Hout, Wout, c.Cout, c.stride, c.pad, sh, K, Kt);
    }
    _Float16* t = bufA; bufA = bufB; bufB = t;
  };
  auto pool = [&](int lvl, int C) {
    long tot = (long)BSZ * HH[lvl + 1] * WW[lvl + 1] * C;
    k_maxpool_f16<<<(unsigned)((tot + 255) / 256), 256, 0, stream>>>(
        bufB, bufA, HH[lvl], WW[lvl], HH[lvl + 1], WW[lvl + 1], C);
    _Float16* t = bufA; bufA = bufB; bufB = t;
  };

  conv(0); conv(1); pool(0, 16);
  conv(2); conv(3); pool(1, 32);
  conv(4); conv(5); pool(2, 64);
  conv(6); conv(7); pool(3, 128);
  conv(8); conv(9); pool(4, 256);
  conv(10);
  conv(11);  // stride-2 VALID, output (B,2,3,256), masked by mptr[6]

  k_linear<<<dim3(BSZ), 128, 0, stream>>>(bufB, linw, linb, outp);
}
